// LSTM_31542239821993
// MI455X (gfx1250) — matrix operand order repaired and checked by reading, not compiled, not run on hardware
//
#include <hip/hip_runtime.h>
#include <hip/hip_bf16.h>
#include <math.h>

// ---------------------------------------------------------------------------
// 2-layer LSTM (B=64, S=512, D=512, H=1024) + FC(1000) for gfx1250 (MI455X).
// bf16 WMMA 16x16x32 with f32 accumulators. A-operand staged in LDS (shared by
// the 8 waves of a block). B-operand (weights, L2-resident) uses a depth-2
// software pipeline with sched_barriers so WMMAs overlap two tile-groups of
// in-flight loads (partial s_wait_loadcnt instead of full stalls).
// ---------------------------------------------------------------------------

typedef __attribute__((ext_vector_type(16))) __bf16 v16bf;
typedef __attribute__((ext_vector_type(8)))  float  v8f;

#define BB   64
#define SS   512
#define DD   512
#define HH   1024
#define OO   1000

__device__ __forceinline__ v8f wmma_bf16(v16bf a, v16bf b, v8f c) {
  // D = A(16x32 bf16) * B(32x16 bf16) + C(16x16 f32)
  return __builtin_amdgcn_wmma_f32_16x16x32_bf16(
      /*neg_a=*/false, a, /*neg_b=*/false, b,
      /*c_mod=*/(short)0, c, /*reuse_a=*/false, /*reuse_b=*/false);
}

__device__ __forceinline__ v16bf ldg16(const __bf16* p) { return *(const v16bf*)p; }

struct BTiles { v16bf i, f, g, o; };

__device__ __forceinline__ BTiles load_btiles(const __bf16* bBase, size_t ntS, int kt) {
  const __bf16* bb = bBase + (size_t)kt * 512;
  BTiles t;
  t.i = ldg16(bb);
  t.f = ldg16(bb + 64 * ntS);
  t.g = ldg16(bb + 128 * ntS);
  t.o = ldg16(bb + 192 * ntS);
  return t;
}

__device__ __forceinline__ void mma_group(v16bf a, const BTiles& b,
                                          v8f& ai, v8f& af, v8f& ag, v8f& ao) {
  ai = wmma_bf16(a, b.i, ai);
  af = wmma_bf16(a, b.f, af);
  ag = wmma_bf16(a, b.g, ag);
  ao = wmma_bf16(a, b.o, ao);
}

// ---------------- packing kernels ----------------
// A 16x32 bf16 tile layout (ISA 7.12.2):
//   lane = (m&15) + 16*((k>>3)&1) ; half = (k&7) + 8*(k>>4)
// B 32x16 bf16 tile layout:
//   lane = (n&15) + 16*((k>>3)&1) ; half = (k&7) + 8*(k>>4)

// Pack x[B,S,D] -> XP[S][mt=4][kt=16][lane=32][half=16] (A layout)
__global__ void pack_x_kernel(const float* __restrict__ x, __bf16* __restrict__ XP) {
  size_t i = (size_t)blockIdx.x * blockDim.x + threadIdx.x;
  if (i >= (size_t)BB * SS * DD) return;
  int d = (int)(i % DD);
  int s = (int)((i / DD) % SS);
  int b = (int)(i / ((size_t)DD * SS));
  float v = x[i];
  int mt = b >> 4, kt = d >> 5, kk = d & 31;
  int lane = (b & 15) + 16 * ((kk >> 3) & 1);
  int half = (kk & 7) + 8 * (kk >> 4);
  XP[(((size_t)(s * 4 + mt) * 16 + kt) * 32 + lane) * 16 + half] = (__bf16)v;
}

// Pack [Wa ; Wb] (rows n of W are B columns; gates = h @ W^T) into B layout:
// BP[ntile][ktTot][lane][half], K = Ka + Kb, rows >= Nvalid padded with 0.
__global__ void pack_w_kernel(const float* __restrict__ Wa, int Ka,
                              const float* __restrict__ Wb, int Kb,
                              int Nrows, int Nvalid, __bf16* __restrict__ BP) {
  int K = Ka + Kb;
  size_t i = (size_t)blockIdx.x * blockDim.x + threadIdx.x;
  if (i >= (size_t)Nrows * K) return;
  int k = (int)(i % K);
  int n = (int)(i / K);
  float v = 0.f;
  if (n < Nvalid)
    v = (k < Ka) ? Wa[(size_t)n * Ka + k] : Wb[(size_t)n * Kb + (k - Ka)];
  int nt = n >> 4, kt = k >> 5, kk = k & 31;
  int lane = (n & 15) + 16 * ((kk >> 3) & 1);
  int half = (kk & 7) + 8 * (kk >> 4);
  int ktTot = K >> 5;
  BP[((size_t)(nt * ktTot + kt) * 32 + lane) * 16 + half] = (__bf16)v;
}

__global__ void bias_sum_kernel(const float* __restrict__ a, const float* __restrict__ b,
                                float* __restrict__ o, int n) {
  int i = blockIdx.x * blockDim.x + threadIdx.x;
  if (i < n) o[i] = a[i] + b[i];
}

// Initialize cell state (f32 [64,1024]) and packed-h (A layout) from h0/c0.
__global__ void init_state_kernel(const float* __restrict__ h0, const float* __restrict__ c0,
                                  float* __restrict__ C0, float* __restrict__ C1,
                                  __bf16* __restrict__ H0Pprev, __bf16* __restrict__ H1Pprev) {
  int i = blockIdx.x * blockDim.x + threadIdx.x;
  if (i >= 2 * BB * HH) return;
  int n = i % HH;
  int m = (i / HH) % BB;
  int l = i / (BB * HH);
  (l ? C1 : C0)[(size_t)m * HH + n] = c0[i];
  float hv = h0[i];
  __bf16* HP = l ? H1Pprev : H0Pprev;
  int mt = m >> 4, kt = n >> 5, kk = n & 31;
  int lane = (m & 15) + 16 * ((kk >> 3) & 1);
  int half = (kk & 7) + 8 * (kk >> 4);
  HP[((size_t)(mt * 32 + kt) * 32 + lane) * 16 + half] = (__bf16)hv;
}

// ---------------- recurrent step ----------------
// One layer, one timestep. gates = [A0|A1] @ BW + bias ; c' = f*c + i*g ; h = o*tanh(c')
// A0/A1: packed A regions ([4 mt][kt0/kt1] tiles) - x|h for L0, h_l0|h_l1 for L1.
// BW: packed B, [256 ntiles][kt0+kt1] tiles, gate order i,f,g,o along N.
// Hout: h written back in packed-A layout ([4 mt][32 kt] tiles) for next consumers.
// Block = 8 waves sharing one M-tile; A slab staged once into LDS.
__global__ __launch_bounds__(256) void lstm_step_kernel(
    const __bf16* __restrict__ A0, int kt0,
    const __bf16* __restrict__ A1, int kt1,
    const __bf16* __restrict__ BW,
    const float* __restrict__ bias,
    float* __restrict__ Cst,
    __bf16* __restrict__ Hout) {
  extern __shared__ __attribute__((aligned(32))) __bf16 sA[];  // [ktTot][32][16]
  const int ktTot = kt0 + kt1;   // even (48 or 64)
  const int tid   = threadIdx.x;
  const int wave  = (blockIdx.x * blockDim.x + tid) >> 5;  // 0..255
  const int lane  = tid & 31;
  const int mt    = wave >> 6;   // 0..3  (M tile; uniform within a block)
  const int ntOut = wave & 63;   // 0..63 (16 output columns of h)

  // ---- stage this block's A slab (both K regions, contiguous per mt) into LDS
  {
    const uint4* s0 = (const uint4*)(A0 + (size_t)mt * kt0 * 512);
    const uint4* s1 = (const uint4*)(A1 + (size_t)mt * kt1 * 512);
    uint4* d0 = (uint4*)sA;                    // kt0*64 uint4
    uint4* d1 = (uint4*)(sA + kt0 * 512);      // kt1*64 uint4
    for (int i = tid; i < kt0 * 64; i += 256) d0[i] = s0[i];
    for (int i = tid; i < kt1 * 64; i += 256) d1[i] = s1[i];
  }
  __syncthreads();

  v8f accI = (v8f){}, accF = (v8f){}, accG = (v8f){}, accO = (v8f){};

  const size_t ntS = (size_t)ktTot * 512;            // halfs per N-tile column
  const __bf16* bBase = BW + (size_t)ntOut * ntS + (size_t)lane * 16;
  const __bf16* sBase = sA + (size_t)lane * 16;

  // ---- depth-2 software-pipelined K loop:
  //   loads for kt+2 / kt+3 are in flight while WMMAs consume kt / kt+1.
  //   sched_barrier(0) pins each load group above the WMMA group so the
  //   scheduler cannot collapse the pipeline back to load->wait(0)->wmma.
  BTiles p0 = load_btiles(bBase, ntS, 0);
  BTiles p1 = load_btiles(bBase, ntS, 1);
  for (int kt = 0; kt < ktTot - 2; kt += 2) {
    BTiles n0 = load_btiles(bBase, ntS, kt + 2);
    __builtin_amdgcn_sched_barrier(0);
    {
      v16bf a = *(const v16bf*)(sBase + (size_t)kt * 512);
      mma_group(a, p0, accI, accF, accG, accO);
    }
    p0 = n0;
    BTiles n1 = load_btiles(bBase, ntS, kt + 3);
    __builtin_amdgcn_sched_barrier(0);
    {
      v16bf a = *(const v16bf*)(sBase + (size_t)(kt + 1) * 512);
      mma_group(a, p1, accI, accF, accG, accO);
    }
    p1 = n1;
  }
  {  // tail: kt = ktTot-2, ktTot-1 (already loaded)
    v16bf a0 = *(const v16bf*)(sBase + (size_t)(ktTot - 2) * 512);
    mma_group(a0, p0, accI, accF, accG, accO);
    v16bf a1 = *(const v16bf*)(sBase + (size_t)(ktTot - 1) * 512);
    mma_group(a1, p1, accI, accF, accG, accO);
  }

  // ---- epilogue: C/D layout -> (m = mt*16 + r + 8*(lane>=16), n = ntOut*16 + lane%16)
  const int n  = (ntOut << 4) + (lane & 15);
  const float bi = bias[n];
  const float bf = bias[HH + n];
  const float bg = bias[2 * HH + n];
  const float bo = bias[3 * HH + n];
  const int mBase = (mt << 4) + ((lane >> 4) << 3);
  // packed-A destination coords for k = n
  const int kt_d = n >> 5;
  const int kk   = n & 31;
  const int laneGrp = 16 * ((kk >> 3) & 1);
  const int half_d  = (kk & 7) + 8 * (kk >> 4);

#pragma unroll
  for (int r = 0; r < 8; ++r) {
    int m = mBase + r;
    float iv = 1.f / (1.f + __expf(-(accI[r] + bi)));
    float fv = 1.f / (1.f + __expf(-(accF[r] + bf)));
    float gv = tanhf(accG[r] + bg);
    float ov = 1.f / (1.f + __expf(-(accO[r] + bo)));
    float c_old = Cst[(size_t)m * HH + n];
    float c_new = fv * c_old + iv * gv;
    Cst[(size_t)m * HH + n] = c_new;
    float hv = ov * tanhf(c_new);
    int lane_d = (m & 15) + laneGrp;
    Hout[((size_t)((m >> 4) * 32 + kt_d) * 32 + lane_d) * 16 + half_d] = (__bf16)hv;
  }
}

// ---------------- FC head ----------------
__global__ __launch_bounds__(256) void fc_kernel(
    const __bf16* __restrict__ A,     // packed h_l1 final, [4 mt][32 kt] tiles
    const __bf16* __restrict__ BFC,   // [64 ntiles][32 kt] tiles (cols>=1000 zero)
    const float* __restrict__ bfc,
    float* __restrict__ out) {
  const int wave = (blockIdx.x * blockDim.x + threadIdx.x) >> 5;
  const int lane = threadIdx.x & 31;
  if (wave >= 4 * 63) return;       // wave-uniform exit keeps EXEC full for WMMA
  const int mt = wave / 63;
  const int nt = wave % 63;
  const __bf16* aBase = A   + ((size_t)mt * 32 * 32 + lane) * 16;
  const __bf16* bBase = BFC + ((size_t)nt * 32 * 32 + lane) * 16;
  v8f acc = (v8f){};
  v16bf a0 = ldg16(aBase), b0 = ldg16(bBase);
  for (int kt = 0; kt < 31; ++kt) {
    v16bf a1 = ldg16(aBase + (size_t)(kt + 1) * 512);
    v16bf b1 = ldg16(bBase + (size_t)(kt + 1) * 512);
    __builtin_amdgcn_sched_barrier(0);
    acc = wmma_bf16(a0, b0, acc);
    a0 = a1; b0 = b1;
  }
  acc = wmma_bf16(a0, b0, acc);
  const int n = (nt << 4) + (lane & 15);
  const int mBase = (mt << 4) + ((lane >> 4) << 3);
  if (n < OO) {
    float bv = bfc[n];
#pragma unroll
    for (int r = 0; r < 8; ++r)
      out[(size_t)(mBase + r) * OO + n] = acc[r] + bv;
  }
}

// ---------------- host launcher ----------------
extern "C" void kernel_launch(void* const* d_in, const int* in_sizes, int n_in,
                              void* d_out, int out_size, void* d_ws, size_t ws_size,
                              hipStream_t stream) {
  (void)in_sizes; (void)n_in; (void)out_size; (void)ws_size;
  const float* x    = (const float*)d_in[0];
  const float* h0   = (const float*)d_in[1];
  const float* c0   = (const float*)d_in[2];
  const float* Wih0 = (const float*)d_in[3];
  const float* Whh0 = (const float*)d_in[4];
  const float* bih0 = (const float*)d_in[5];
  const float* bhh0 = (const float*)d_in[6];
  const float* Wih1 = (const float*)d_in[7];
  const float* Whh1 = (const float*)d_in[8];
  const float* bih1 = (const float*)d_in[9];
  const float* bhh1 = (const float*)d_in[10];
  const float* Wfc  = (const float*)d_in[11];
  const float* bfc  = (const float*)d_in[12];

  char* ws = (char*)d_ws;
  size_t off = 0;
  auto alloc = [&](size_t bytes) -> void* {
    void* p = ws + off;
    off += (bytes + 255) & ~(size_t)255;
    return p;
  };

  __bf16* XP   = (__bf16*)alloc((size_t)SS * 4 * 16 * 32 * 16 * 2);  // 33.5 MB
  __bf16* BW0  = (__bf16*)alloc((size_t)256 * 48 * 512 * 2);         // 12.6 MB
  __bf16* BW1  = (__bf16*)alloc((size_t)256 * 64 * 512 * 2);         // 16.8 MB
  __bf16* BFC  = (__bf16*)alloc((size_t)64  * 32 * 512 * 2);         //  2.0 MB
  __bf16* h0p[2] = { (__bf16*)alloc(4 * 32 * 512 * 2), (__bf16*)alloc(4 * 32 * 512 * 2) };
  __bf16* h1p[2] = { (__bf16*)alloc(4 * 32 * 512 * 2), (__bf16*)alloc(4 * 32 * 512 * 2) };
  float* C0    = (float*)alloc((size_t)BB * HH * 4);
  float* C1    = (float*)alloc((size_t)BB * HH * 4);
  float* bias0 = (float*)alloc(4 * HH * 4);
  float* bias1 = (float*)alloc(4 * HH * 4);

  // Pack weights (bf16, B layout, gate-major N): [Wih^T ; Whh^T] K-concat.
  {
    size_t n = (size_t)4096 * 1536;
    pack_w_kernel<<<(unsigned)((n + 255) / 256), 256, 0, stream>>>(Wih0, 512, Whh0, 1024, 4096, 4096, BW0);
  }
  {
    size_t n = (size_t)4096 * 2048;
    pack_w_kernel<<<(unsigned)((n + 255) / 256), 256, 0, stream>>>(Wih1, 1024, Whh1, 1024, 4096, 4096, BW1);
  }
  {
    size_t n = (size_t)1024 * 1024;
    pack_w_kernel<<<(unsigned)((n + 255) / 256), 256, 0, stream>>>(Wfc, 1024, nullptr, 0, 1024, OO, BFC);
  }
  bias_sum_kernel<<<(4 * HH + 255) / 256, 256, 0, stream>>>(bih0, bhh0, bias0, 4 * HH);
  bias_sum_kernel<<<(4 * HH + 255) / 256, 256, 0, stream>>>(bih1, bhh1, bias1, 4 * HH);

  // Pack x (all timesteps) into A layout.
  {
    size_t n = (size_t)BB * SS * DD;
    pack_x_kernel<<<(unsigned)((n + 255) / 256), 256, 0, stream>>>(x, XP);
  }
  // Initialize c state + packed h(-1) into parity-1 buffers (step 0 reads parity 1).
  init_state_kernel<<<(2 * BB * HH + 255) / 256, 256, 0, stream>>>(h0, c0, C0, C1, h0p[1], h1p[1]);

  // Recurrent chain: stream ordering provides the step/layer dependency.
  for (int s = 0; s < SS; ++s) {
    const int cur = s & 1, prev = cur ^ 1;
    const __bf16* xs = XP + (size_t)s * 4 * 16 * 32 * 16;  // per-step packed x
    // layer 0: gates = [x_s | h0_{s-1}] @ [Wih0^T ; Whh0^T]   (K = 512+1024, 48 KB LDS)
    lstm_step_kernel<<<32, 256, 48 * 1024, stream>>>(xs, 16, h0p[prev], 32, BW0, bias0, C0, h0p[cur]);
    // layer 1: gates = [h0_s | h1_{s-1}] @ [Wih1^T ; Whh1^T]  (K = 1024+1024, 64 KB LDS)
    lstm_step_kernel<<<32, 256, 64 * 1024, stream>>>(h0p[cur], 32, h1p[prev], 32, BW1, bias1, C1, h1p[cur]);
  }

  // out = h1[S-1] @ Wfc^T + bfc
  fc_kernel<<<32, 256, 0, stream>>>(h1p[(SS - 1) & 1], BFC, bfc, (float*)d_out);
}
